// Pointnet2Backbone_4836133175856
// MI455X (gfx1250) — matrix operand order repaired and checked by reading, not compiled, not run on hardware
//
#include <hip/hip_runtime.h>
#include <hip/hip_bf16.h>

typedef __bf16 v16bf __attribute__((ext_vector_type(16)));
typedef __bf16 v8bf  __attribute__((ext_vector_type(8)));
typedef float  v8f   __attribute__((ext_vector_type(8)));

#define BATCH 2
#define NPTS  20000
#define BN_EPSF 1e-5f

// ---------------------------------------------------------------------------
// Farthest point sampling: one block per batch, sequential iterations,
// block-parallel min-dist update + argmax reduction in LDS.
// ---------------------------------------------------------------------------
__global__ void fps_kernel(const float* __restrict__ xyz, int* __restrict__ out_idx,
                           float* __restrict__ mind_ws, int n, int npoint) {
  const int b = blockIdx.x;
  const float* p = xyz + (long)b * n * 3;
  int* out = out_idx + (long)b * npoint;
  float* mind = mind_ws + (long)b * n;
  __shared__ float s_val[256];
  __shared__ int   s_idx[256];
  __shared__ int   s_last;
  const int t = threadIdx.x;
  for (int i = t; i < n; i += blockDim.x) mind[i] = 1e10f;
  if (t == 0) { out[0] = 0; s_last = 0; }
  __syncthreads();
  for (int it = 1; it < npoint; ++it) {
    const int last = s_last;
    const float lx = p[last * 3 + 0], ly = p[last * 3 + 1], lz = p[last * 3 + 2];
    float best = -1.0f; int bi = n - 1;
    for (int i = t; i < n; i += blockDim.x) {
      const float dx = p[i * 3 + 0] - lx;
      const float dy = p[i * 3 + 1] - ly;
      const float dz = p[i * 3 + 2] - lz;
      float md = fminf(mind[i], dx * dx + dy * dy + dz * dz);
      mind[i] = md;
      if (md > best) { best = md; bi = i; }
    }
    s_val[t] = best; s_idx[t] = bi;
    __syncthreads();
    for (int s = blockDim.x >> 1; s > 0; s >>= 1) {
      if (t < s) {
        const float ov = s_val[t + s]; const int oi = s_idx[t + s];
        if (ov > s_val[t] || (ov == s_val[t] && oi < s_idx[t])) { s_val[t] = ov; s_idx[t] = oi; }
      }
      __syncthreads();
    }
    if (t == 0) { out[it] = s_idx[0]; s_last = s_idx[0]; }
    __syncthreads();
  }
}

__global__ void gather_xyz_kernel(const float* __restrict__ xyz, const int* __restrict__ idx,
                                  float* __restrict__ out, int n, int m) {
  const int i = blockIdx.x * blockDim.x + threadIdx.x;
  if (i >= BATCH * m) return;
  const int b = i / m;
  const int j = idx[i];
  const float* p = xyz + ((long)b * n + j) * 3;
  out[i * 3 + 0] = p[0]; out[i * 3 + 1] = p[1]; out[i * 3 + 2] = p[2];
}

// Ball query: first nsample in-radius points in index order.
__global__ void ball_query_kernel(const float* __restrict__ xyz, const float* __restrict__ centers,
                                  int* __restrict__ idx, int n, int m, float radius, int nsample) {
  const int gid = blockIdx.x * blockDim.x + threadIdx.x;
  if (gid >= BATCH * m) return;
  const int b = gid / m;
  const float* p = xyz + (long)b * n * 3;
  const float cx = centers[gid * 3 + 0], cy = centers[gid * 3 + 1], cz = centers[gid * 3 + 2];
  int* o = idx + (long)gid * nsample;
  const float r2 = radius * radius;
  int cnt = 0; int first = n - 1;
  for (int j = 0; j < n && cnt < nsample; ++j) {
    const float dx = cx - p[j * 3 + 0];
    const float dy = cy - p[j * 3 + 1];
    const float dz = cz - p[j * 3 + 2];
    if (dx * dx + dy * dy + dz * dz <= r2) {
      if (cnt == 0) first = j;
      o[cnt++] = j;
    }
  }
  for (; cnt < nsample; ++cnt) o[cnt] = first;
}

// Grouped features in bf16, row stride K32 (zero-padded): [ (xyz-c)/r , feats ]
__global__ void group_kernel(const float* __restrict__ xyz, const float* __restrict__ centers,
                             const float* __restrict__ feats, const int* __restrict__ idx,
                             __bf16* __restrict__ out, int n, int m, int nsample, int cf,
                             int K32, float inv_radius) {
  const long gid = (long)blockIdx.x * blockDim.x + threadIdx.x;
  const long total = (long)BATCH * m * nsample;
  if (gid >= total) return;
  const int b = (int)(gid / ((long)m * nsample));
  const long rem = gid % ((long)m * nsample);
  const int c = (int)(rem / nsample);
  const int j = idx[gid];
  const float* p  = xyz + ((long)b * n + j) * 3;
  const float* ce = centers + ((long)b * m + c) * 3;
  __bf16* o = out + gid * K32;
  o[0] = (__bf16)((p[0] - ce[0]) * inv_radius);
  o[1] = (__bf16)((p[1] - ce[1]) * inv_radius);
  o[2] = (__bf16)((p[2] - ce[2]) * inv_radius);
  for (int k = 0; k < cf; ++k) o[3 + k] = (__bf16)feats[((long)b * n + j) * cf + k];
  for (int k = 3 + cf; k < K32; ++k) o[k] = (__bf16)0.0f;
}

// Pack f32 weights (Cout,Cin) -> bf16 (Cout,K32), zero-padded K.
__global__ void pack_w_kernel(const float* __restrict__ W, __bf16* __restrict__ Wp,
                              int Cout, int Cin, int K32) {
  const long i = (long)blockIdx.x * blockDim.x + threadIdx.x;
  if (i >= (long)Cout * K32) return;
  const int r = (int)(i / K32);
  const int c = (int)(i % K32);
  Wp[i] = (c < Cin) ? (__bf16)W[(long)r * Cin + c] : (__bf16)0.0f;
}

// ---------------------------------------------------------------------------
// WMMA bf16 linear: Y(E,Cout) = X(E,K32-bf16) * Wp(Cout,K32-bf16)^T.
// One wave per block computes a 16x16 f32 tile. Fragments loaded straight
// from global with b128 loads in the ISA 7.12.2 register layout; no LDS,
// no predication (E and Cout are multiples of 16, K padded to 32).
// ---------------------------------------------------------------------------
__global__ __launch_bounds__(32) void linear_wmma_kernel(
    const __bf16* __restrict__ X, const __bf16* __restrict__ Wp,
    float* __restrict__ Y, int E, int K32, int Cout) {
  const int lane = threadIdx.x & 31;
  const int row0 = blockIdx.x << 4;
  const int col0 = blockIdx.y << 4;
  const int m  = lane & 15;
  const int hi = lane >> 4;  // 0 or 1
  // A: lanes 0-15 hold rows M, K chunks [0..7],[16..23]; lanes 16-31: [8..15],[24..31]
  const __bf16* ap = X  + (long)(row0 + m) * K32 + hi * 8;
  // B: lane -> column N; lanes 0-15 K[0..7],[16..23]; lanes 16-31: [8..15],[24..31]
  const __bf16* bp = Wp + (long)(col0 + m) * K32 + hi * 8;
  v8f acc = {};
  for (int k0 = 0; k0 < K32; k0 += 32) {
    if (k0 + 32 < K32) {
      __builtin_prefetch(ap + k0 + 32, 0, 1);
      __builtin_prefetch(bp + k0 + 32, 0, 1);
    }
    const v8bf a0 = *(const v8bf*)(ap + k0);
    const v8bf a1 = *(const v8bf*)(ap + k0 + 16);
    const v8bf b0 = *(const v8bf*)(bp + k0);
    const v8bf b1 = *(const v8bf*)(bp + k0 + 16);
    const v16bf a = __builtin_shufflevector(a0, a1, 0, 1, 2, 3, 4, 5, 6, 7,
                                            8, 9, 10, 11, 12, 13, 14, 15);
    const v16bf b = __builtin_shufflevector(b0, b1, 0, 1, 2, 3, 4, 5, 6, 7,
                                            8, 9, 10, 11, 12, 13, 14, 15);
    acc = __builtin_amdgcn_wmma_f32_16x16x32_bf16(false, a, false, b,
                                                  (short)0, acc, false, false);
  }
  // D: lane = N, VGPR r = M (+8 for lanes 16-31)
  const int mbase = hi << 3;
#pragma unroll
  for (int r = 0; r < 8; ++r)
    Y[(long)(row0 + mbase + r) * Cout + col0 + m] = acc[r];
}

// Per-channel sum / sumsq: one block per channel, LDS tree reduction.
__global__ void bn_stats_kernel(const float* __restrict__ Y, int E, int C,
                                float* __restrict__ stats) {
  const int c = blockIdx.x;
  const int t = threadIdx.x;
  float s = 0.0f, sq = 0.0f;
  for (int e = t; e < E; e += blockDim.x) {
    const float v = Y[(long)e * C + c];
    s += v; sq += v * v;
  }
  __shared__ float sh1[256], sh2[256];
  sh1[t] = s; sh2[t] = sq;
  __syncthreads();
  for (int st = 128; st > 0; st >>= 1) {
    if (t < st) { sh1[t] += sh1[t + st]; sh2[t] += sh2[t + st]; }
    __syncthreads();
  }
  if (t == 0) { stats[c] = sh1[0]; stats[C + c] = sh2[0]; }
}

// BN + ReLU: write bf16 activations for next WMMA layer; optional f32 copy.
__global__ void bn_relu_kernel(const float* __restrict__ Y, __bf16* __restrict__ Xb,
                               float* __restrict__ Xf, const float* __restrict__ g,
                               const float* __restrict__ bt, const float* __restrict__ stats,
                               long total, int E, int C) {
  const long i = (long)blockIdx.x * blockDim.x + threadIdx.x;
  if (i >= total) return;
  const int c = (int)(i % C);
  const float inv_e = 1.0f / (float)E;
  const float mean = stats[c] * inv_e;
  const float var = stats[C + c] * inv_e - mean * mean;
  const float x = (Y[i] - mean) * rsqrtf(var + BN_EPSF) * g[c] + bt[c];
  const float r = fmaxf(x, 0.0f);
  Xb[i] = (__bf16)r;
  if (Xf) Xf[i] = r;
}

__global__ void maxpool_kernel(const __bf16* __restrict__ X, float* __restrict__ out,
                               int rows, int nsample, int C) {
  const long i = (long)blockIdx.x * blockDim.x + threadIdx.x;
  if (i >= (long)rows * C) return;
  const long r = i / C;
  const int c = (int)(i % C);
  float m = -1e30f;
  for (int s = 0; s < nsample; ++s)
    m = fmaxf(m, (float)X[((long)r * nsample + s) * C + c]);
  out[i] = m;
}

// 3-NN inverse-distance interpolation -> bf16 columns [0,C2) of concat buffer.
__global__ void three_interp_kernel(const float* __restrict__ xyz1, const float* __restrict__ xyz2,
                                    const float* __restrict__ f2, __bf16* __restrict__ out,
                                    int n1, int n2, int C2, int stride) {
  const int gid = blockIdx.x * blockDim.x + threadIdx.x;
  if (gid >= BATCH * n1) return;
  const int b = gid / n1;
  const float* p = xyz1 + (long)gid * 3;
  const float* q = xyz2 + (long)b * n2 * 3;
  float d0 = 1e30f, d1 = 1e30f, d2 = 1e30f;
  int i0 = 0, i1 = 0, i2 = 0;
  for (int j = 0; j < n2; ++j) {
    const float dx = q[j * 3 + 0] - p[0];
    const float dy = q[j * 3 + 1] - p[1];
    const float dz = q[j * 3 + 2] - p[2];
    const float dd = dx * dx + dy * dy + dz * dz;
    if (dd < d0)      { d2 = d1; i2 = i1; d1 = d0; i1 = i0; d0 = dd; i0 = j; }
    else if (dd < d1) { d2 = d1; i2 = i1; d1 = dd; i1 = j; }
    else if (dd < d2) { d2 = dd; i2 = j; }
  }
  float w0 = 1.0f / (d0 + 1e-8f), w1 = 1.0f / (d1 + 1e-8f), w2 = 1.0f / (d2 + 1e-8f);
  const float winv = 1.0f / (w0 + w1 + w2);
  w0 *= winv; w1 *= winv; w2 *= winv;
  const float* fb = f2 + (long)b * n2 * C2;
  __bf16* o = out + (long)gid * stride;
  for (int c = 0; c < C2; ++c)
    o[c] = (__bf16)(w0 * fb[(long)i0 * C2 + c] + w1 * fb[(long)i1 * C2 + c] +
                    w2 * fb[(long)i2 * C2 + c]);
}

__global__ void concat_f1_kernel(const float* __restrict__ f1, __bf16* __restrict__ out,
                                 int n1, int C1, int C2, int stride) {
  const long i = (long)blockIdx.x * blockDim.x + threadIdx.x;
  if (i >= (long)BATCH * n1 * C1) return;
  const long r = i / C1;
  const int c = (int)(i % C1);
  out[r * stride + C2 + c] = (__bf16)f1[i];
}

__global__ void inds_kernel(const int* __restrict__ fps1, float* __restrict__ out,
                            int take, int full) {
  const int i = blockIdx.x * blockDim.x + threadIdx.x;
  if (i >= BATCH * take) return;
  const int b = i / take, k = i % take;
  out[i] = (float)fps1[b * full + k];
}

// ---------------------------------------------------------------------------
extern "C" void kernel_launch(void* const* d_in, const int* in_sizes, int n_in,
                              void* d_out, int out_size, void* d_ws, size_t ws_size,
                              hipStream_t stream) {
  (void)in_sizes; (void)n_in; (void)out_size; (void)ws_size;
  const float* pc = (const float*)d_in[0];
  auto saW = [&](int m, int l) { return (const float*)d_in[1 + m * 9 + l * 3 + 0]; };
  auto saG = [&](int m, int l) { return (const float*)d_in[1 + m * 9 + l * 3 + 1]; };
  auto saB = [&](int m, int l) { return (const float*)d_in[1 + m * 9 + l * 3 + 2]; };
  auto fpW = [&](int m, int l) { return (const float*)d_in[37 + m * 6 + l * 3 + 0]; };
  auto fpG = [&](int m, int l) { return (const float*)d_in[37 + m * 6 + l * 3 + 1]; };
  auto fpB = [&](int m, int l) { return (const float*)d_in[37 + m * 6 + l * 3 + 2]; };
  auto c32 = [](int x) { return (x + 31) & ~31; };

  // --- workspace bump allocator ---
  char* base = (char*)d_ws;
  size_t off = 0;
  auto alloc = [&](size_t bytes) -> void* {
    off = (off + 255) & ~(size_t)255;
    void* p = base + off;
    off += bytes;
    return p;
  };
  float* mind = (float*)alloc((size_t)BATCH * NPTS * 4);
  int* fps1 = (int*)alloc((size_t)BATCH * 1024 * 4);
  int* fps2 = (int*)alloc((size_t)BATCH * 512 * 4);
  int* fps3 = (int*)alloc((size_t)BATCH * 256 * 4);
  int* fps4 = (int*)alloc((size_t)BATCH * 64 * 4);
  float* xyz1 = (float*)alloc((size_t)BATCH * 1024 * 3 * 4);
  float* xyz2 = (float*)alloc((size_t)BATCH * 512 * 3 * 4);
  float* xyz3 = (float*)alloc((size_t)BATCH * 256 * 3 * 4);
  float* xyz4 = (float*)alloc((size_t)BATCH * 64 * 3 * 4);
  int* idx1 = (int*)alloc((size_t)BATCH * 1024 * 32 * 4);
  int* idx2 = (int*)alloc((size_t)BATCH * 512 * 32 * 4);
  int* idx3 = (int*)alloc((size_t)BATCH * 256 * 16 * 4);
  int* idx4 = (int*)alloc((size_t)BATCH * 64 * 32 * 4);
  float* sa1f = (float*)alloc((size_t)BATCH * 1024 * 64 * 4);
  float* sa2f = (float*)alloc((size_t)BATCH * 512 * 128 * 4);
  float* sa3f = (float*)alloc((size_t)BATCH * 256 * 256 * 4);
  float* sa4f = (float*)alloc((size_t)BATCH * 64 * 256 * 4);
  float* fp1f = (float*)alloc((size_t)BATCH * 256 * 256 * 4);
  float* stats = (float*)alloc(2 * 256 * 4);
  const size_t ACT_ELEMS = 4456448;  // >= 65536 x 64
  __bf16* bufX = (__bf16*)alloc(ACT_ELEMS * 2);   // bf16 activations (padded K)
  float*  bufY = (float*)alloc(ACT_ELEMS * 4);    // f32 pre-BN linear output
  __bf16* wpack = (__bf16*)alloc((size_t)256 * 512 * 2);

  // One shared-MLP layer: pack W -> WMMA linear -> stats -> BN+ReLU.
  auto mlp_layer = [&](const __bf16* X, const float* W, const float* g, const float* b,
                       int E, int Cin, int Cout, __bf16* Xnext, float* Xf32) {
    const int K32 = c32(Cin);
    const long ptot = (long)Cout * K32;
    pack_w_kernel<<<(unsigned)((ptot + 255) / 256), 256, 0, stream>>>(W, wpack, Cout, Cin, K32);
    dim3 grid(E / 16, Cout / 16);
    linear_wmma_kernel<<<grid, 32, 0, stream>>>(X, wpack, bufY, E, K32, Cout);
    bn_stats_kernel<<<Cout, 256, 0, stream>>>(bufY, E, Cout, stats);
    const long total = (long)E * Cout;
    bn_relu_kernel<<<(unsigned)((total + 255) / 256), 256, 0, stream>>>(
        bufY, Xnext, Xf32, g, b, stats, total, E, Cout);
  };

  auto run_sa = [&](const float* xyz_in, const float* feats_in, int n, int npoint,
                    float radius, int nsample, int cf, int mod, const int* couts,
                    float* new_xyz, float* feats_out, int* fps_out, int* idx_buf) {
    fps_kernel<<<BATCH, 256, 0, stream>>>(xyz_in, fps_out, mind, n, npoint);
    gather_xyz_kernel<<<(BATCH * npoint + 255) / 256, 256, 0, stream>>>(
        xyz_in, fps_out, new_xyz, n, npoint);
    ball_query_kernel<<<(BATCH * npoint + 255) / 256, 256, 0, stream>>>(
        xyz_in, new_xyz, idx_buf, n, npoint, radius, nsample);
    const long gtotal = (long)BATCH * npoint * nsample;
    const int K32g = c32(3 + cf);
    group_kernel<<<(unsigned)((gtotal + 255) / 256), 256, 0, stream>>>(
        xyz_in, new_xyz, feats_in, idx_buf, bufX, n, npoint, nsample, cf, K32g,
        1.0f / radius);
    const int E = BATCH * npoint * nsample;
    int cin = 3 + cf;
    for (int l = 0; l < 3; ++l) {
      const int cout = couts[l];
      mlp_layer(bufX, saW(mod, l), saG(mod, l), saB(mod, l), E, cin, cout, bufX, nullptr);
      cin = cout;
    }
    const long ptotal = (long)BATCH * npoint * cin;
    maxpool_kernel<<<(unsigned)((ptotal + 255) / 256), 256, 0, stream>>>(
        bufX, feats_out, BATCH * npoint, nsample, cin);
  };

  auto run_fp = [&](const float* x1, const float* x2, const float* f1, const float* f2,
                    int n1, int n2, int C1, int C2, int mod, const int* couts,
                    float* out) {
    const int stride = C1 + C2;  // 512 or 384: multiples of 32, no padding
    three_interp_kernel<<<(BATCH * n1 + 255) / 256, 256, 0, stream>>>(
        x1, x2, f2, bufX, n1, n2, C2, stride);
    const long ctotal = (long)BATCH * n1 * C1;
    concat_f1_kernel<<<(unsigned)((ctotal + 255) / 256), 256, 0, stream>>>(
        f1, bufX, n1, C1, C2, stride);
    const int E = BATCH * n1;
    int cin = stride;
    for (int l = 0; l < 2; ++l) {
      const int cout = couts[l];
      float* f32dst = (l == 1) ? out : nullptr;  // final layer also lands in f32 target
      mlp_layer(bufX, fpW(mod, l), fpG(mod, l), fpB(mod, l), E, cin, cout, bufX, f32dst);
      cin = cout;
    }
  };

  static const int sa_couts[4][3] = {{32, 32, 64}, {64, 64, 128}, {128, 128, 256}, {128, 128, 256}};
  static const int fp_couts[2][2] = {{256, 256}, {128, 256}};

  run_sa(pc,   nullptr, NPTS, 1024, 0.1f, 32, 0,   0, sa_couts[0], xyz1, sa1f, fps1, idx1);
  run_sa(xyz1, sa1f,    1024, 512,  0.2f, 32, 64,  1, sa_couts[1], xyz2, sa2f, fps2, idx2);
  run_sa(xyz2, sa2f,    512,  256,  0.4f, 16, 128, 2, sa_couts[2], xyz3, sa3f, fps3, idx3);
  run_sa(xyz3, sa3f,    256,  64,   0.1f, 32, 256, 3, sa_couts[3], xyz4, sa4f, fps4, idx4);

  float* outF = (float*)d_out;
  run_fp(xyz3, xyz4, sa3f, sa4f, 256, 64,  256, 256, 0, fp_couts[0], fp1f);
  run_fp(xyz2, xyz3, sa2f, fp1f, 512, 256, 128, 256, 1, fp_couts[1], outF);

  // Tail outputs: sa2_xyz then fp2_inds (= sa1 fps idx[:, :512]) as floats.
  float* out_xyz = outF + (size_t)BATCH * 512 * 256;
  float* out_ind = out_xyz + (size_t)BATCH * 512 * 3;
  hipMemcpyAsync(out_xyz, xyz2, (size_t)BATCH * 512 * 3 * 4, hipMemcpyDeviceToDevice, stream);
  inds_kernel<<<(BATCH * 512 + 255) / 256, 256, 0, stream>>>(fps1, out_ind, 512, 1024);
}
